// SparseMoE_31628139167808
// MI455X (gfx1250) — matrix operand order repaired and checked by reading, not compiled, not run on hardware
//
#include <hip/hip_runtime.h>
#include <hip/hip_bf16.h>
#include <math.h>

// ---- problem constants (match reference) ----
#define DDIM 1024
#define HDIM 2048
#define NEXP 8
#define TOPK 2
#define TTOK 2048              // B*S = 2*1024
#define SLOTS (TTOK * TOPK)    // 4096 routed (token,expert) pairs

typedef __attribute__((ext_vector_type(16))) __bf16 v16bf;
typedef __attribute__((ext_vector_type(8)))  float  v8f;
typedef __hip_bfloat16 bf16;

__device__ __forceinline__ float gelu_exact(float x) {
  return 0.5f * x * (1.0f + erff(x * 0.70710678118654752f));
}

// ---------------------------------------------------------------------------
// 0. zero per-expert counters
// ---------------------------------------------------------------------------
__global__ void k_zero_counts(int* __restrict__ counts) {
  if (threadIdx.x < NEXP) counts[threadIdx.x] = 0;
}

// ---------------------------------------------------------------------------
// 1. router: logits -> softmax -> top-2 -> counts
//    block = 64 threads, handles 8 tokens; thread = (token, expert)
// ---------------------------------------------------------------------------
__global__ void k_router(const float* __restrict__ flat,
                         const float* __restrict__ rw,
                         const float* __restrict__ rb,
                         int* __restrict__ topk_idx, float* __restrict__ topk_w,
                         int* __restrict__ counts) {
  __shared__ float lg[64];
  const int tid = threadIdx.x;
  const int tl  = tid >> 3;          // token-in-block 0..7
  const int e   = tid & 7;           // expert 0..7
  const int t   = blockIdx.x * 8 + tl;
  const float* x = flat + (size_t)t * DDIM;
  float acc = rb[e];
  for (int d = 0; d < DDIM; ++d) acc += x[d] * rw[d * NEXP + e];
  lg[tid] = acc;
  __syncthreads();
  if (e == 0) {
    const float* l = lg + tl * 8;
    float mx = l[0];
    for (int i = 1; i < 8; ++i) mx = fmaxf(mx, l[i]);
    float p[8], s = 0.f;
    for (int i = 0; i < 8; ++i) { p[i] = __expf(l[i] - mx); s += p[i]; }
    const float inv = 1.0f / s;
    for (int i = 0; i < 8; ++i) p[i] *= inv;
    int i1 = 0;
    for (int i = 1; i < 8; ++i) if (p[i] > p[i1]) i1 = i;
    int i2 = (i1 == 0) ? 1 : 0;
    for (int i = 0; i < 8; ++i) if (i != i1 && p[i] > p[i2]) i2 = i;
    topk_idx[t * 2 + 0] = i1;  topk_idx[t * 2 + 1] = i2;
    topk_w[t * 2 + 0]  = p[i1]; topk_w[t * 2 + 1]  = p[i2];
    atomicAdd(&counts[i1], 1);
    atomicAdd(&counts[i2], 1);
  }
}

// ---------------------------------------------------------------------------
// 2a. exclusive scan over 8 counts -> offsets; init scatter cursors
// ---------------------------------------------------------------------------
__global__ void k_offsets(const int* __restrict__ counts,
                          int* __restrict__ offsets, int* __restrict__ cursor) {
  if (threadIdx.x == 0 && blockIdx.x == 0) {
    int off = 0;
    for (int e = 0; e < NEXP; ++e) {
      offsets[e] = off; cursor[e] = off; off += counts[e];
    }
    offsets[NEXP] = off;
  }
}

// 2b. scatter token ids into compact per-expert slot lists
__global__ void k_scatter(const int* __restrict__ topk_idx,
                          int* __restrict__ cursor,
                          int* __restrict__ slot_map, int* __restrict__ slot_token) {
  const int t = blockIdx.x * blockDim.x + threadIdx.x;
  if (t >= TTOK) return;
  for (int k = 0; k < TOPK; ++k) {
    const int e   = topk_idx[t * 2 + k];
    const int pos = atomicAdd(&cursor[e], 1);
    slot_map[t * 2 + k] = pos;
    slot_token[pos]     = t;
  }
}

// ---------------------------------------------------------------------------
// 3a. fp32 -> bf16 activation convert
// ---------------------------------------------------------------------------
__global__ void k_cvt(const float* __restrict__ src, bf16* __restrict__ dst, int n) {
  const int i = blockIdx.x * blockDim.x + threadIdx.x;
  if (i < n) dst[i] = __float2bfloat16(src[i]);
}

// 3b. tiled transpose + fp32->bf16: in [R,C] -> out [C,R]
//     blockIdx.z = 0 -> shared matrix, 1..8 -> routed expert z-1
__global__ void k_transpose(const float* __restrict__ s0, const float* __restrict__ sE,
                            bf16* __restrict__ d0, bf16* __restrict__ dE,
                            int R, int C) {
  __shared__ float tile[32][33];
  const int z = blockIdx.z;
  const float* src = (z == 0) ? s0 : sE + (size_t)(z - 1) * R * C;
  bf16*        dst = (z == 0) ? d0 : dE + (size_t)(z - 1) * R * C;
  const int x  = blockIdx.x * 32 + threadIdx.x;   // input col
  const int y0 = blockIdx.y * 32;                 // input row base
  for (int j = threadIdx.y; j < 32; j += 8)
    tile[j][threadIdx.x] = src[(size_t)(y0 + j) * C + x];
  __syncthreads();
  const int ox  = y0 + threadIdx.x;               // output col (R index)
  const int oyb = blockIdx.x * 32;                // output row base (C index)
  for (int j = threadIdx.y; j < 32; j += 8)
    dst[(size_t)(oyb + j) * R + ox] = __float2bfloat16(tile[threadIdx.x][j]);
}

// ---------------------------------------------------------------------------
// 4a. FFN1: H = gelu(X @ W1 + b1), bf16 out.  W1t is [(e),HDIM,DDIM] (N-major).
//     block = 256 threads = 8 waves; wave -> 16x16 f32 accumulator tile.
//     routed: rows gathered through slot_token, per-expert range from offsets/counts.
// ---------------------------------------------------------------------------
__global__ void k_ffn1(const bf16* __restrict__ X, const bf16* __restrict__ W,
                       const float* __restrict__ b1, bf16* __restrict__ Hout,
                       const int* __restrict__ slot_token,
                       const int* __restrict__ offsets, const int* __restrict__ counts,
                       int routed) {
  const int lane = threadIdx.x & 31;
  const int wave = threadIdx.x >> 5;
  const int lh   = lane >> 4;          // which 16-lane half
  const int l16  = lane & 15;
  const int n0   = blockIdx.x * 128 + wave * 16;
  const int e      = routed ? (int)blockIdx.z : 0;
  const int base   = routed ? offsets[e] : 0;
  const int mcount = routed ? counts[e]  : TTOK;
  const int m0     = blockIdx.y * 16;
  if (m0 >= mcount) return;
  int mr = m0 + l16; if (mr >= mcount) mr = mcount - 1;   // clamp tail rows
  const int arow = routed ? slot_token[base + mr] : mr;
  const unsigned* Ap = (const unsigned*)(X + (size_t)arow * DDIM);
  const unsigned* Wp = (const unsigned*)(W + ((size_t)e * HDIM + n0 + l16) * DDIM);
  v8f acc = {};
  #pragma unroll 2
  for (int kk = 0; kk < DDIM / 32; ++kk) {
    union { unsigned u[8]; v16bf v; } a, b;
    const int kb = kk * 16;                       // dword base of this K=32 chunk
    #pragma unroll
    for (int r = 0; r < 8; ++r) {
      // ISA 16-bit A 16x32 layout: lanes0-15 V0-3:K0-7,V4-7:K16-23; lanes16-31 +8
      a.u[r] = Ap[kb + 4 * lh + (r < 4 ? r : 4 + r)];
      // B 32x16: same K layout as A (V0-3:K0-7,V4-7:K16-23; lanes16-31 +8), N = lane%16
      b.u[r] = Wp[kb + 4 * lh + (r < 4 ? r : 4 + r)];
    }
    __builtin_prefetch((const void*)(Wp + kb + 16), 0, 1);
    acc = __builtin_amdgcn_wmma_f32_16x16x32_bf16(false, a.v, false, b.v,
                                                  (short)0, acc, false, false);
  }
  const int col = n0 + l16;
  const float bias = b1[(size_t)e * HDIM + col];
  #pragma unroll
  for (int j = 0; j < 8; ++j) {                   // C layout: row = j + 8*lh
    const int mrel = m0 + j + 8 * lh;
    if (mrel < mcount)
      Hout[(size_t)(base + mrel) * HDIM + col] =
          __float2bfloat16(gelu_exact(acc[j] + bias));
  }
}

// ---------------------------------------------------------------------------
// 4b. FFN2: Out = H @ W2 (+ b2 for shared), f32 out. W2t is [(e),DDIM,HDIM].
// ---------------------------------------------------------------------------
__global__ void k_ffn2(const bf16* __restrict__ Hin, const bf16* __restrict__ W,
                       const float* __restrict__ b2, float* __restrict__ Out,
                       const int* __restrict__ offsets, const int* __restrict__ counts,
                       int routed) {
  const int lane = threadIdx.x & 31;
  const int wave = threadIdx.x >> 5;
  const int lh   = lane >> 4;
  const int l16  = lane & 15;
  const int n0   = blockIdx.x * 128 + wave * 16;
  const int e      = routed ? (int)blockIdx.z : 0;
  const int base   = routed ? offsets[e] : 0;
  const int mcount = routed ? counts[e]  : TTOK;
  const int m0     = blockIdx.y * 16;
  if (m0 >= mcount) return;
  int mr = m0 + l16; if (mr >= mcount) mr = mcount - 1;
  const unsigned* Ap = (const unsigned*)(Hin + (size_t)(base + mr) * HDIM);
  const unsigned* Wp = (const unsigned*)(W + ((size_t)e * DDIM + n0 + l16) * HDIM);
  v8f acc = {};
  #pragma unroll 2
  for (int kk = 0; kk < HDIM / 32; ++kk) {
    union { unsigned u[8]; v16bf v; } a, b;
    const int kb = kk * 16;
    #pragma unroll
    for (int r = 0; r < 8; ++r) {
      a.u[r] = Ap[kb + 4 * lh + (r < 4 ? r : 4 + r)];
      b.u[r] = Wp[kb + 4 * lh + (r < 4 ? r : 4 + r)];
    }
    __builtin_prefetch((const void*)(Wp + kb + 16), 0, 1);
    acc = __builtin_amdgcn_wmma_f32_16x16x32_bf16(false, a.v, false, b.v,
                                                  (short)0, acc, false, false);
  }
  const int col = n0 + l16;
  const float bias = routed ? 0.0f : b2[col];
  #pragma unroll
  for (int j = 0; j < 8; ++j) {
    const int mrel = m0 + j + 8 * lh;
    if (mrel < mcount)
      Out[(size_t)(base + mrel) * DDIM + col] = acc[j] + bias;
  }
}

// ---------------------------------------------------------------------------
// 5. mix: out[t,d] += sum_k w_k * (eo[slot(t,k), d] + rb2[e_k, d])
// ---------------------------------------------------------------------------
__global__ void k_mix(float* __restrict__ out, const float* __restrict__ eo,
                      const float* __restrict__ rb2,
                      const int* __restrict__ topk_idx,
                      const float* __restrict__ topk_w,
                      const int* __restrict__ slot_map) {
  const int idx = blockIdx.x * blockDim.x + threadIdx.x;  // over T*D
  const int t = idx >> 10;          // / DDIM
  const int d = idx & (DDIM - 1);   // % DDIM
  float acc = out[idx];
  #pragma unroll
  for (int k = 0; k < TOPK; ++k) {
    const int   e    = topk_idx[t * 2 + k];
    const float w    = topk_w[t * 2 + k];
    const int   slot = slot_map[t * 2 + k];
    acc += w * (eo[(size_t)slot * DDIM + d] + rb2[(size_t)e * DDIM + d]);
  }
  out[idx] = acc;
}

// ---------------------------------------------------------------------------
// host launch
// ---------------------------------------------------------------------------
extern "C" void kernel_launch(void* const* d_in, const int* in_sizes, int n_in,
                              void* d_out, int out_size, void* d_ws, size_t ws_size,
                              hipStream_t stream) {
  const float* hidden   = (const float*)d_in[0];
  const float* router_w = (const float*)d_in[1];
  const float* router_b = (const float*)d_in[2];
  const float* sw1      = (const float*)d_in[3];
  const float* sb1      = (const float*)d_in[4];
  const float* sw2      = (const float*)d_in[5];
  const float* sb2      = (const float*)d_in[6];
  const float* rw1      = (const float*)d_in[7];
  const float* rb1      = (const float*)d_in[8];
  const float* rw2      = (const float*)d_in[9];
  const float* rb2      = (const float*)d_in[10];
  float* out = (float*)d_out;

  // ---- carve workspace ----
  char* p = (char*)d_ws;
  auto take = [&](size_t bytes) -> void* {
    void* r = (void*)p;
    p += (bytes + 255) & ~(size_t)255;
    return r;
  };
  int*   counts     = (int*)take(NEXP * 4);
  int*   offsets    = (int*)take((NEXP + 1) * 4);
  int*   cursor     = (int*)take(NEXP * 4);
  int*   topk_idx   = (int*)take((size_t)TTOK * 2 * 4);
  float* topk_w     = (float*)take((size_t)TTOK * 2 * 4);
  int*   slot_map   = (int*)take((size_t)TTOK * 2 * 4);
  int*   slot_token = (int*)take((size_t)SLOTS * 4);
  bf16*  Xb         = (bf16*)take((size_t)TTOK * DDIM * 2);
  bf16*  w1ts       = (bf16*)take((size_t)HDIM * DDIM * 2);
  bf16*  w1tr       = (bf16*)take((size_t)NEXP * HDIM * DDIM * 2);
  bf16*  w2ts       = (bf16*)take((size_t)DDIM * HDIM * 2);
  bf16*  w2tr       = (bf16*)take((size_t)NEXP * DDIM * HDIM * 2);
  bf16*  hs         = (bf16*)take((size_t)TTOK * HDIM * 2);
  bf16*  hr         = (bf16*)take((size_t)SLOTS * HDIM * 2);
  float* eo         = (float*)take((size_t)SLOTS * DDIM * 4);

  // 0-2: routing
  k_zero_counts<<<1, 32, 0, stream>>>(counts);
  k_router<<<TTOK / 8, 64, 0, stream>>>(hidden, router_w, router_b,
                                        topk_idx, topk_w, counts);
  k_offsets<<<1, 1, 0, stream>>>(counts, offsets, cursor);
  k_scatter<<<TTOK / 256, 256, 0, stream>>>(topk_idx, cursor, slot_map, slot_token);

  // 3: precision conversion / weight transposition (bf16, [N,K] layout)
  k_cvt<<<(TTOK * DDIM) / 256, 256, 0, stream>>>(hidden, Xb, TTOK * DDIM);
  {
    dim3 tb(32, 8);
    dim3 g1(HDIM / 32, DDIM / 32, NEXP + 1);  // W1: [D,H] -> [H,D]
    k_transpose<<<g1, tb, 0, stream>>>(sw1, rw1, w1ts, w1tr, DDIM, HDIM);
    dim3 g2(DDIM / 32, HDIM / 32, NEXP + 1);  // W2: [H,D] -> [D,H]
    k_transpose<<<g2, tb, 0, stream>>>(sw2, rw2, w2ts, w2tr, HDIM, DDIM);
  }

  // 4: WMMA GEMMs
  {
    dim3 b(256);
    dim3 g1s(HDIM / 128, TTOK / 16, 1);
    k_ffn1<<<g1s, b, 0, stream>>>(Xb, w1ts, sb1, hs, nullptr, nullptr, nullptr, 0);
    dim3 g1r(HDIM / 128, TTOK / 16, NEXP);
    k_ffn1<<<g1r, b, 0, stream>>>(Xb, w1tr, rb1, hr, slot_token, offsets, counts, 1);
    dim3 g2s(DDIM / 128, TTOK / 16, 1);
    k_ffn2<<<g2s, b, 0, stream>>>(hs, w2ts, sb2, out, nullptr, nullptr, 0);
    dim3 g2r(DDIM / 128, TTOK / 16, NEXP);
    k_ffn2<<<g2r, b, 0, stream>>>(hr, w2tr, nullptr, eo, offsets, counts, 1);
  }

  // 5: weighted mix of routed expert outputs into d_out
  k_mix<<<(TTOK * DDIM) / 256, 256, 0, stream>>>(out, eo, rb2,
                                                 topk_idx, topk_w, slot_map);
}
